// MatrixAttention_15530601742895
// MI455X (gfx1250) — matrix operand order repaired and checked by reading, not compiled, not run on hardware
//
#include <hip/hip_runtime.h>

// ---------------------------------------------------------------------------
// Cosine-similarity batched GEMM for MI455X (gfx1250, wave32, WMMA).
//   C[b,i,j] = <l2norm(m1[b,i,:]), l2norm(m2[b,j,:])>
// B=8, R=2048, D=128.  Strategy: normalize rows once into bf16 hi/lo split
// (near-fp32 precision), then tensor-core GEMM with V_WMMA_F32_16X16X32_BF16.
// ---------------------------------------------------------------------------

typedef __attribute__((ext_vector_type(16))) __bf16 v16bf;
typedef __attribute__((ext_vector_type(8)))  __bf16 v8bf;
typedef __attribute__((ext_vector_type(8)))  float  v8f;
typedef __attribute__((ext_vector_type(4)))  float  v4f;

#define BATCH 8
#define RDIM  2048
#define DDIM  128
#define ROWS_PER_MAT (BATCH * RDIM)                     // 16384 rows per matrix
#define MAT_ELEMS    ((size_t)ROWS_PER_MAT * DDIM)      // 2,097,152 elements

// Round-to-nearest-even float32 -> bf16 bit pattern, plus residual split.
__device__ __forceinline__ void split_bf16(float x, unsigned short& h, unsigned short& l) {
    unsigned u  = __float_as_uint(x);
    unsigned hb = (u + 0x7FFFu + ((u >> 16) & 1u)) & 0xFFFF0000u;   // RNE truncate
    h = (unsigned short)(hb >> 16);
    float r = x - __uint_as_float(hb);                               // exact residual
    unsigned ur = __float_as_uint(r);
    l = (unsigned short)((ur + 0x7FFFu + ((ur >> 16) & 1u)) >> 16);  // RNE again
}

// ---------------------------------------------------------------------------
// Kernel 1: L2-normalize each row (one wave per 128-float row) and emit
// bf16 hi/lo split into workspace. 32 lanes x 4 floats = 128 = D.
// ---------------------------------------------------------------------------
__global__ __launch_bounds__(256) void nrm_split_kernel(const float* __restrict__ m1,
                                                        const float* __restrict__ m2,
                                                        unsigned short* __restrict__ ws) {
    unsigned short* hi1 = ws;
    unsigned short* lo1 = ws + MAT_ELEMS;
    unsigned short* hi2 = ws + 2 * MAT_ELEMS;
    unsigned short* lo2 = ws + 3 * MAT_ELEMS;

    const int lane = threadIdx.x & 31;
    const int wv   = threadIdx.x >> 5;
    const int row  = blockIdx.x * 8 + wv;               // 0 .. 32767 (both matrices)

    const float* src;
    unsigned short* dh;
    unsigned short* dl;
    if (row < ROWS_PER_MAT) {
        src = m1 + (size_t)row * DDIM;
        dh  = hi1 + (size_t)row * DDIM;
        dl  = lo1 + (size_t)row * DDIM;
    } else {
        const int r = row - ROWS_PER_MAT;
        src = m2 + (size_t)r * DDIM;
        dh  = hi2 + (size_t)r * DDIM;
        dl  = lo2 + (size_t)r * DDIM;
    }

    v4f x = *(const v4f*)(src + lane * 4);
    float ss = x.x * x.x + x.y * x.y + x.z * x.z + x.w * x.w;
#pragma unroll
    for (int m = 16; m >= 1; m >>= 1) ss += __shfl_xor(ss, m, 32);   // wave32 reduce

    const float s = rsqrtf(fmaxf(ss, 1e-8f));   // matches reference: x*rsqrt(max(ss,eps))

    float xs[4] = {x.x * s, x.y * s, x.z * s, x.w * s};
    unsigned short h[4], l[4];
#pragma unroll
    for (int i = 0; i < 4; ++i) split_bf16(xs[i], h[i], l[i]);

    // 8-byte packed stores (lane*4 bf16 = 8B aligned)
    uint2 hp, lp;
    hp.x = (unsigned)h[0] | ((unsigned)h[1] << 16);
    hp.y = (unsigned)h[2] | ((unsigned)h[3] << 16);
    lp.x = (unsigned)l[0] | ((unsigned)l[1] << 16);
    lp.y = (unsigned)l[2] | ((unsigned)l[3] << 16);
    *(uint2*)(dh + lane * 4) = hp;
    *(uint2*)(dl + lane * 4) = lp;
}

// ---------------------------------------------------------------------------
// Kernel 2: tiled WMMA GEMM. Each 256-thread block (8 waves) computes a
// 128x128 tile of C for one batch. Wave w owns a 16-row M-strip and all
// 8 N-subtiles; K=128 is consumed in 4 steps of 32 with
// v_wmma_f32_16x16x32_bf16, 3 WMMAs per subtile for the hi/lo product terms.
// ---------------------------------------------------------------------------
__global__ __launch_bounds__(256) void cos_gemm_kernel(const unsigned short* __restrict__ ws,
                                                       float* __restrict__ out) {
    const unsigned short* hi1 = ws;
    const unsigned short* lo1 = ws + MAT_ELEMS;
    const unsigned short* hi2 = ws + 2 * MAT_ELEMS;
    const unsigned short* lo2 = ws + 3 * MAT_ELEMS;

    const int lane = threadIdx.x & 31;
    const int wave = threadIdx.x >> 5;
    const int half = lane >> 4;      // 0: lanes 0-15, 1: lanes 16-31
    const int l16  = lane & 15;

    const int b     = blockIdx.z;
    const int Mbase = blockIdx.y * 128 + wave * 16;
    const int Nbase = blockIdx.x * 128;

    const unsigned short* Ah = hi1 + ((size_t)b * RDIM + Mbase) * DDIM;
    const unsigned short* Al = lo1 + ((size_t)b * RDIM + Mbase) * DDIM;
    const unsigned short* Bh = hi2 + ((size_t)b * RDIM + Nbase) * DDIM;
    const unsigned short* Bl = lo2 + ((size_t)b * RDIM + Nbase) * DDIM;

    v8f acc[8];
    const v8f vzero = {0.f, 0.f, 0.f, 0.f, 0.f, 0.f, 0.f, 0.f};
#pragma unroll
    for (int n = 0; n < 8; ++n) acc[n] = vzero;

    union Frag { v16bf v; v8bf h[2]; };

#pragma unroll
    for (int ks = 0; ks < 4; ++ks) {
        const int k0 = ks * 32;

        // A fragment (contiguous k order, same as B): row M = l16.
        //   lanes 0-15 : K 0..15 contiguous
        //   lanes16-31 : K 16..31 contiguous
        const unsigned short* arh = Ah + (size_t)l16 * DDIM + k0 + half * 16;
        const unsigned short* arl = Al + (size_t)l16 * DDIM + k0 + half * 16;
        Frag a_hi, a_lo;
        a_hi.v = *(const v16bf*)(arh);
        a_lo.v = *(const v16bf*)(arl);

#pragma unroll
        for (int n = 0; n < 8; ++n) {
            // B fragment (contiguous k order, same as A): column N = l16 of subtile n.
            //   lanes 0-15 hold K k0..k0+15, lanes 16-31 hold K k0+16..k0+31,
            //   i.e. one contiguous 32B run of the row-major normalized m2 row.
            const unsigned short* brh = Bh + (size_t)(n * 16 + l16) * DDIM + k0 + half * 16;
            const unsigned short* brl = Bl + (size_t)(n * 16 + l16) * DDIM + k0 + half * 16;
            const v16bf b_hi = *(const v16bf*)(brh);
            const v16bf b_lo = *(const v16bf*)(brl);

            acc[n] = __builtin_amdgcn_wmma_f32_16x16x32_bf16(
                false, a_hi.v, false, b_hi, (short)0, acc[n], false, false);
            acc[n] = __builtin_amdgcn_wmma_f32_16x16x32_bf16(
                false, a_hi.v, false, b_lo, (short)0, acc[n], false, false);
            acc[n] = __builtin_amdgcn_wmma_f32_16x16x32_bf16(
                false, a_lo.v, false, b_hi, (short)0, acc[n], false, false);
        }
    }

    // C/D layout (32-bit, 16x16): lanes 0-15 N=lane, VGPR r -> M=r;
    //                             lanes 16-31 N=lane-16, VGPR r -> M=8+r.
    float* outp = out + ((size_t)b * RDIM + Mbase + half * 8) * RDIM + Nbase + l16;
#pragma unroll
    for (int n = 0; n < 8; ++n) {
#pragma unroll
        for (int r = 0; r < 8; ++r) {
            outp[(size_t)r * RDIM + n * 16] = acc[n][r];
        }
    }
}

// ---------------------------------------------------------------------------
extern "C" void kernel_launch(void* const* d_in, const int* in_sizes, int n_in,
                              void* d_out, int out_size, void* d_ws, size_t ws_size,
                              hipStream_t stream) {
    const float* m1 = (const float*)d_in[0];
    const float* m2 = (const float*)d_in[1];
    float* out = (float*)d_out;
    unsigned short* ws = (unsigned short*)d_ws;   // needs 4 * MAT_ELEMS * 2B = 16 MB

    // 32768 rows total, 8 waves (rows) per block -> 4096 blocks.
    nrm_split_kernel<<<4096, 256, 0, stream>>>(m1, m2, ws);

    dim3 grid(RDIM / 128, RDIM / 128, BATCH);     // (tileN, tileM, batch)
    cos_gemm_kernel<<<grid, 256, 0, stream>>>(ws, out);
}
